// MultiHeadAttention_my_42099269435883
// MI455X (gfx1250) — compile-verified, matrix operand order checked by reading, not run on hardware
//
#include <hip/hip_runtime.h>

#define DM   1024
#define SEQ  2048
#define BS   2
#define NH   16
#define DK   64
#define ROWS (BS * SEQ)   // 4096

// Toggle: emit gfx1250 async global->LDS copies via inline asm (ASYNCcnt path).
// Set to 0 to fall back to plain load + ds_store staging.
#define USE_ASYNC_LDS_ASM 1

typedef __attribute__((ext_vector_type(16))) __bf16 v16bf;
typedef __attribute__((ext_vector_type(8)))  __bf16 v8bf;
typedef __attribute__((ext_vector_type(8)))  float  v8f;
typedef __attribute__((ext_vector_type(4)))  float  v4f;

// ---------- helpers ----------

static __device__ inline __bf16 f2bf(float f) {
    union { float f; unsigned u; } a; a.f = f;
    unsigned r = a.u + 0x7fffu + ((a.u >> 16) & 1u);   // round-to-nearest-even
    union { unsigned short s; __bf16 b; } o; o.s = (unsigned short)(r >> 16);
    return o.b;
}

static __device__ inline v8f zero8() {
    v8f z = {0.f, 0.f, 0.f, 0.f, 0.f, 0.f, 0.f, 0.f};
    return z;
}

// A/B 16-bit fragment (16x32 / 32x16): per ISA layout, lane L holds row/col
// (L&15); lanes 0-15 carry K {0..7, 16..23}, lanes 16-31 carry K {8..15, 24..31}.
// p points at element (row, kbase) of a row-major matrix. Pure 16B loads.
static __device__ inline v16bf load_frag_bf(const __bf16* p, int lane) {
    int k0 = (lane & 16) ? 8 : 0;
    v8bf lo = *(const v8bf*)(p + k0);
    v8bf hi = *(const v8bf*)(p + k0 + 16);
    v16bf r;
#pragma unroll
    for (int i = 0; i < 8; i++) { r[i] = lo[i]; r[i + 8] = hi[i]; }
    return r;
}

static __device__ inline v8f wmma_bf16(v16bf a, v16bf b, v8f c) {
    return __builtin_amdgcn_wmma_f32_16x16x32_bf16(
        /*neg_a=*/false, a, /*neg_b=*/false, b,
        /*c_mod=*/(short)0, c, /*reuse_a=*/false, /*reuse_b=*/false);
}

// 16-byte global -> LDS copy. Async path: per ISA aperture rules the low 32
// bits of a generic pointer into LDS are the LDS byte address, so we can feed
// the instruction's VDST directly without address-space gymnastics.
static __device__ inline void cp16_g2l(const __bf16* g, __bf16* l) {
#if USE_ASYNC_LDS_ASM
    unsigned loff = (unsigned)(size_t)l;                    // LDS byte offset
    unsigned long long ga = (unsigned long long)(size_t)g;  // global address
    asm volatile("global_load_async_to_lds_b128 %0, %1, off"
                 :: "v"(loff), "v"(ga)
                 : "memory");
#else
    *(v8bf*)l = *(const v8bf*)g;
#endif
}

static __device__ inline void wait_async_copies() {
#if USE_ASYNC_LDS_ASM
    asm volatile("s_wait_asynccnt 0" ::: "memory");
#endif
}

// ---------- kernel 0: one-time fp32 -> bf16 convert (bandwidth-bound) ----------
__global__ __launch_bounds__(256) void mha_cvt_kernel(
    const float* __restrict__ src, __bf16* __restrict__ dst, int n8)
{
    int t = blockIdx.x * blockDim.x + threadIdx.x;
    if (t >= n8) return;
    const v4f* s = (const v4f*)(src + (size_t)t * 8);
    v4f a = s[0];
    v4f b = s[1];
    v8bf o;
#pragma unroll
    for (int i = 0; i < 4; i++) { o[i] = f2bf(a[i]); o[i + 4] = f2bf(b[i]); }
    *(v8bf*)(dst + (size_t)t * 8) = o;
}

// ---------- kernel 1: projection Y = X * W^T + b  (all-bf16 WMMA) ----------
// mode 0: store row-major bf16 [ROWS][DM]     (Q, K)
// mode 1: store per-head transposed bf16 Vt[b][h][dk][seq]   (V)
__global__ __launch_bounds__(256) void mha_proj_kernel(
    const __bf16* __restrict__ X, const __bf16* __restrict__ W,
    const float* __restrict__ bias,
    __bf16* __restrict__ out_rm, __bf16* __restrict__ out_vt, int mode)
{
    int lane = threadIdx.x & 31;
    int l15  = lane & 15;
    int wid  = (blockIdx.x * blockDim.x + threadIdx.x) >> 5;
    int mtile = wid >> 4;        // 0..255  (16 rows each)
    int ngrp  = wid & 15;        // 0..15   (64 cols each)
    int mbase = mtile * 16;
    int nbase = ngrp * 64;
    int arow  = mbase + l15;

    v8f acc[4];
#pragma unroll
    for (int f = 0; f < 4; f++) acc[f] = zero8();

    for (int k = 0; k < DM; k += 32) {
        v16bf a = load_frag_bf(X + (size_t)arow * DM + k, lane);
#pragma unroll
        for (int f = 0; f < 4; f++) {
            v16bf b = load_frag_bf(W + (size_t)(nbase + f * 16 + l15) * DM + k, lane);
            acc[f] = wmma_bf16(a, b, acc[f]);
        }
    }

    int mo = (lane & 16) ? 8 : 0;
    if (mode == 0) {
#pragma unroll
        for (int f = 0; f < 4; f++) {
            int n = nbase + f * 16 + l15;
            float bb = bias[n];
#pragma unroll
            for (int r = 0; r < 8; r++)
                out_rm[(size_t)(mbase + mo + r) * DM + n] = f2bf(acc[f][r] + bb);
        }
    } else {
        int b = mbase >> 11;                 // batch
        int s = (mbase & (SEQ - 1)) + mo;    // seq position of first row
#pragma unroll
        for (int f = 0; f < 4; f++) {
            int ng = nbase + f * 16 + l15;
            int h = ng >> 6, dl = ng & 63;
            float bb = bias[ng];
            v8bf pk;
#pragma unroll
            for (int r = 0; r < 8; r++) pk[r] = f2bf(acc[f][r] + bb);
            *(v8bf*)(out_vt + (size_t)b * DM * SEQ + (size_t)(h * DK + dl) * SEQ + s) = pk;
        }
    }
}

// ---------- kernel 2: flash attention ----------
// 8 waves/block all work on the same (b,h): K/V tiles are staged cooperatively
// into LDS once per 32-key step (async global->LDS) and all waves build their
// WMMA fragments from LDS. Rows padded to kill bank conflicts.
#define KT_STRIDE 72   // 64 + 8 pad  (36 dwords -> conflict-free fragment reads)
#define VT_STRIDE 40   // 32 + 8 pad  (20 dwords)
#define PT_STRIDE 40   // 32 + 8 pad

__global__ __launch_bounds__(256) void mha_attn_kernel(
    const __bf16* __restrict__ qb, const __bf16* __restrict__ kb,
    const __bf16* __restrict__ vt, __bf16* __restrict__ at)
{
    __shared__ __bf16 plds[8][16 * PT_STRIDE];   // per-wave P-tile scratch
    __shared__ __bf16 ktile[32 * KT_STRIDE];     // keys x d_k  (4.6 KB)
    __shared__ __bf16 vtile[64 * VT_STRIDE];     // d_k x keys  (5.0 KB)

    int tid   = threadIdx.x;
    int lane  = tid & 31;
    int l15   = lane & 15;
    int wslot = tid >> 5;
    int wid   = (blockIdx.x * blockDim.x + tid) >> 5;
    int qt = wid & 127;           // q tile within (b,h); 8 consecutive per block
    int bh = wid >> 7;            // block-uniform
    int b  = bh >> 4, h = bh & 15;
    int qbase = qt * 16;

    const __bf16* Qb = qb + (size_t)b * SEQ * DM + h * DK;
    const __bf16* Kb = kb + (size_t)b * SEQ * DM + h * DK;
    const __bf16* Vb = vt + (size_t)b * DM * SEQ + (size_t)(h * DK) * SEQ;

    // staging addresses for this thread (8 bf16 per thread per tile)
    int krow = tid >> 3, kcol = (tid & 7) * 8;     // K: 32 x 64
    int vrow = tid >> 2, vcol = (tid & 3) * 8;     // V: 64 x 32

    // Q fragments for the whole tile (d_k = 64 -> two K=32 chunks), in VGPRs
    v16bf qf0 = load_frag_bf(Qb + (size_t)(qbase + l15) * DM + 0,  lane);
    v16bf qf1 = load_frag_bf(Qb + (size_t)(qbase + l15) * DM + 32, lane);

    v8f o[4];
#pragma unroll
    for (int f = 0; f < 4; f++) o[f] = zero8();
    float mrow[8], lrow[8];
#pragma unroll
    for (int r = 0; r < 8; r++) { mrow[r] = -1e30f; lrow[r] = 0.f; }

    for (int k2 = 0; k2 < SEQ; k2 += 32) {
        // ---- cooperative stage of K (32x64) and V^T (64x32) tiles ----
        cp16_g2l(Kb + (size_t)(k2 + krow) * DM + kcol, &ktile[krow * KT_STRIDE + kcol]);
        cp16_g2l(Vb + (size_t)vrow * SEQ + k2 + vcol,  &vtile[vrow * VT_STRIDE + vcol]);
        wait_async_copies();
        __syncthreads();

        // ---- S = Q * K^T for 32 keys (two 16-col tiles), fragments from LDS ----
        v8f s0 = zero8(), s1 = zero8();
        {
            v16bf b00 = load_frag_bf(&ktile[l15 * KT_STRIDE + 0],         lane);
            v16bf b10 = load_frag_bf(&ktile[(16 + l15) * KT_STRIDE + 0],  lane);
            s0 = wmma_bf16(qf0, b00, s0);
            s1 = wmma_bf16(qf0, b10, s1);
            v16bf b01 = load_frag_bf(&ktile[l15 * KT_STRIDE + 32],        lane);
            v16bf b11 = load_frag_bf(&ktile[(16 + l15) * KT_STRIDE + 32], lane);
            s0 = wmma_bf16(qf1, b01, s0);
            s1 = wmma_bf16(qf1, b11, s1);
        }

        // ---- online softmax (row stats across the 16-lane half-groups) ----
#pragma unroll
        for (int r = 0; r < 8; r++) {
            float x0 = s0[r] * 0.125f;          // 1/sqrt(64)
            float x1 = s1[r] * 0.125f;
            float mx = fmaxf(x0, x1);
            mx = fmaxf(mx, __shfl_xor(mx, 1, 32));
            mx = fmaxf(mx, __shfl_xor(mx, 2, 32));
            mx = fmaxf(mx, __shfl_xor(mx, 4, 32));
            mx = fmaxf(mx, __shfl_xor(mx, 8, 32));
            float nm = fmaxf(mrow[r], mx);
            float al = __expf(mrow[r] - nm);
            float p0 = __expf(x0 - nm);
            float p1 = __expf(x1 - nm);
            float ps = p0 + p1;
            ps += __shfl_xor(ps, 1, 32);
            ps += __shfl_xor(ps, 2, 32);
            ps += __shfl_xor(ps, 4, 32);
            ps += __shfl_xor(ps, 8, 32);
            lrow[r] = lrow[r] * al + ps;
            mrow[r] = nm;
#pragma unroll
            for (int f = 0; f < 4; f++) o[f][r] *= al;
            // D-layout -> LDS (row m, 32 key cols) for A-layout reload
            int m = r + ((lane & 16) ? 8 : 0);
            plds[wslot][m * PT_STRIDE + l15]      = f2bf(p0);
            plds[wslot][m * PT_STRIDE + 16 + l15] = f2bf(p1);
        }

        asm volatile("s_wait_dscnt 0" ::: "memory");

        // ---- O += P(16x32) * V(32x64), V fragments from LDS ----
        v16bf pa = load_frag_bf(&plds[wslot][l15 * PT_STRIDE], lane);
#pragma unroll
        for (int f = 0; f < 4; f++) {
            v16bf vbf = load_frag_bf(&vtile[(f * 16 + l15) * VT_STRIDE], lane);
            o[f] = wmma_bf16(pa, vbf, o[f]);
        }

        __syncthreads();   // protect tiles before next stage
    }

    // ---- normalize and store concat-layout bf16 attn output ----
    int mo = (lane & 16) ? 8 : 0;
#pragma unroll
    for (int f = 0; f < 4; f++) {
        int n = h * DK + f * 16 + l15;
#pragma unroll
        for (int r = 0; r < 8; r++) {
            float inv = 1.0f / lrow[r];
            at[((size_t)b * SEQ + qbase + mo + r) * DM + n] = f2bf(o[f][r] * inv);
        }
    }
}

// ---------- kernel 3: output projection out = attn * W_o^T + b_o (fp32 out) ----------
__global__ __launch_bounds__(256) void mha_outproj_kernel(
    const __bf16* __restrict__ A, const __bf16* __restrict__ W,
    const float* __restrict__ bias, float* __restrict__ out)
{
    int lane = threadIdx.x & 31;
    int l15  = lane & 15;
    int wid  = (blockIdx.x * blockDim.x + threadIdx.x) >> 5;
    int mtile = wid >> 4;
    int ngrp  = wid & 15;
    int mbase = mtile * 16;
    int nbase = ngrp * 64;
    int arow  = mbase + l15;

    v8f acc[4];
#pragma unroll
    for (int f = 0; f < 4; f++) acc[f] = zero8();

    for (int k = 0; k < DM; k += 32) {
        v16bf a = load_frag_bf(A + (size_t)arow * DM + k, lane);
#pragma unroll
        for (int f = 0; f < 4; f++) {
            v16bf b = load_frag_bf(W + (size_t)(nbase + f * 16 + l15) * DM + k, lane);
            acc[f] = wmma_bf16(a, b, acc[f]);
        }
    }

    int mo = (lane & 16) ? 8 : 0;
#pragma unroll
    for (int f = 0; f < 4; f++) {
        int n = nbase + f * 16 + l15;
        float bb = bias[n];
#pragma unroll
        for (int r = 0; r < 8; r++)
            out[(size_t)(mbase + mo + r) * DM + n] = acc[f][r] + bb;
    }
}

// ---------- host launcher ----------
extern "C" void kernel_launch(void* const* d_in, const int* in_sizes, int n_in,
                              void* d_out, int out_size, void* d_ws, size_t ws_size,
                              hipStream_t stream) {
    const float* q  = (const float*)d_in[0];
    const float* k  = (const float*)d_in[1];
    const float* v  = (const float*)d_in[2];
    const float* Wq = (const float*)d_in[3];
    const float* bq = (const float*)d_in[4];
    const float* Wk = (const float*)d_in[5];
    const float* bk = (const float*)d_in[6];
    const float* Wv = (const float*)d_in[7];
    const float* bv = (const float*)d_in[8];
    const float* Wo = (const float*)d_in[9];
    const float* bo = (const float*)d_in[10];
    float* out = (float*)d_out;

    char* ws = (char*)d_ws;
    const size_t ACT = (size_t)ROWS * DM * sizeof(unsigned short);  // 8 MB
    __bf16* qb = (__bf16*)(ws);             // Q projected, bf16 row-major
    __bf16* kb = (__bf16*)(ws + ACT);       // K projected
    __bf16* vt = (__bf16*)(ws + 2 * ACT);   // V projected, per-head transposed
    __bf16* at = (__bf16*)(ws + 3 * ACT);   // attention output (concat layout)
    __bf16* xb = (__bf16*)(ws + 4 * ACT);   // staged bf16 activation (reused)
    __bf16* wb = (__bf16*)(ws + 5 * ACT);   // staged bf16 weight (reused)

    dim3 blk(256);
    dim3 grd(512);                       // 4096 waves
    const int actN8 = (ROWS * DM) / 8;
    const int wgtN8 = (DM * DM) / 8;
    dim3 grdA((actN8 + 255) / 256);
    dim3 grdW((wgtN8 + 255) / 256);

    // Q = q Wq^T : convert once, GEMM in pure bf16
    mha_cvt_kernel<<<grdA, blk, 0, stream>>>(q, xb, actN8);
    mha_cvt_kernel<<<grdW, blk, 0, stream>>>(Wq, wb, wgtN8);
    mha_proj_kernel<<<grd, blk, 0, stream>>>(xb, wb, bq, qb, nullptr, 0);
    // K
    mha_cvt_kernel<<<grdA, blk, 0, stream>>>(k, xb, actN8);
    mha_cvt_kernel<<<grdW, blk, 0, stream>>>(Wk, wb, wgtN8);
    mha_proj_kernel<<<grd, blk, 0, stream>>>(xb, wb, bk, kb, nullptr, 0);
    // V (stored transposed per head)
    mha_cvt_kernel<<<grdA, blk, 0, stream>>>(v, xb, actN8);
    mha_cvt_kernel<<<grdW, blk, 0, stream>>>(Wv, wb, wgtN8);
    mha_proj_kernel<<<grd, blk, 0, stream>>>(xb, wb, bv, nullptr, vt, 1);
    // flash attention
    mha_attn_kernel<<<grd, blk, 0, stream>>>(qb, kb, vt, at);
    // output projection
    mha_cvt_kernel<<<grdW, blk, 0, stream>>>(Wo, wb, wgtN8);
    mha_outproj_kernel<<<grd, blk, 0, stream>>>(at, wb, bo, out);
}
